// MoEBitNetFFN_56332791054921
// MI455X (gfx1250) — 8-bit matrix operand order repaired and checked by reading, compiled, not run on hardware
//
#include <hip/hip_runtime.h>
#include <hip/hip_bf16.h>
#include <stdint.h>


typedef __attribute__((ext_vector_type(8))) int v8i;

static constexpr int kN = 4096;
static constexpr int kD = 1024;
static constexpr int kF = 4096;
static constexpr int kE = 16;
static constexpr int kMaxRows = kN * 2 + kE * 32;
static constexpr int kMaxTiles = kMaxRows / 16;
static constexpr int kMGroups = kMaxRows / 32;
static constexpr float kRmsEps = 1e-6f;

__device__ __forceinline__ float gelu_tanh(float x) {
  const float c = 0.7978845608028654f;
  return 0.5f * x * (1.0f + tanhf(c * (x + 0.044715f * x * x * x)));
}

__device__ __forceinline__ v8i load_a_tile(const int8_t* qa, int row,
                                           int kdim, int k0, int half) {
  const int8_t* p = qa + (size_t)row * kdim + k0 + half * 16;
  v8i a;
  int2 t0 = *(const int2*)(p + 0);  a[0] = t0.x; a[1] = t0.y;
  int2 t1 = *(const int2*)(p + 8); a[2] = t1.x; a[3] = t1.y;
  int2 t2 = *(const int2*)(p + 32); a[4] = t2.x; a[5] = t2.y;
  int2 t3 = *(const int2*)(p + 40); a[6] = t3.x; a[7] = t3.y;
  return a;
}

__device__ __forceinline__ v8i load_b_tile(const int8_t* qw, int nrow,
                                           int kdim, int k0, int half) {
  const int8_t* p = qw + (size_t)nrow * kdim + k0 + half * 16;
  int4 b0 = *(const int4*)(p + 0);
  int4 b1 = *(const int4*)(p + 32);
  v8i b;
  b[0] = b0.x; b[1] = b0.y; b[2] = b0.z; b[3] = b0.w;
  b[4] = b1.x; b[5] = b1.y; b[6] = b1.z; b[7] = b1.w;
  return b;
}

__global__ void k_zero_out(float* out) {
  size_t i = (size_t)blockIdx.x * blockDim.x + threadIdx.x;
  if (i < (size_t)kN * kD) out[i] = 0.0f;
}

__global__ void k_init_misc(int* perm, float* pweight, int* cnt, int* fill,
                            float* p_sum, float* f_cnt, float* wsum) {
  int i = blockIdx.x * blockDim.x + threadIdx.x;
  if (i < kMaxRows) { perm[i] = -1; pweight[i] = 0.0f; }
  if (i < kE) { cnt[i] = 0; fill[i] = 0; p_sum[i] = 0.0f; f_cnt[i] = 0.0f; }
  if (i < 2 * kE) wsum[i] = 0.0f;
}

__global__ void k_weight_sum(const float* w1, const float* w2, float* wsum) {
  int e = blockIdx.y, z = blockIdx.z;
  const float* w = (z ? w2 : w1) + (size_t)e * kF * kD;
  size_t nel = (size_t)kF * kD;
  float s = 0.0f;
  for (size_t i = (size_t)blockIdx.x * blockDim.x + threadIdx.x; i < nel;
       i += (size_t)gridDim.x * blockDim.x)
    s += fabsf(w[i]);
  __shared__ float red[256];
  red[threadIdx.x] = s;
  __syncthreads();
  for (int w2_ = 128; w2_ > 0; w2_ >>= 1) {
    if (threadIdx.x < w2_) red[threadIdx.x] += red[threadIdx.x + w2_];
    __syncthreads();
  }
  if (threadIdx.x == 0) atomicAdd(&wsum[z * kE + e], red[0]);
}

__global__ void k_router(const float* x, const float* rw, int* topi,
                         float* topw, int* cnt, float* p_sum, float* f_cnt) {
  int t = blockIdx.x;
  int e = threadIdx.x >> 5, lane = threadIdx.x & 31;
  const float* xr = x + (size_t)t * kD;
  const float* wr = rw + (size_t)e * kD;
  float p = 0.0f;
  for (int d = lane; d < kD; d += 32) p += xr[d] * wr[d];
  for (int o = 16; o > 0; o >>= 1) p += __shfl_down(p, o, 32);
  __shared__ float lg[kE];
  if (lane == 0) lg[e] = p;
  __syncthreads();
  if (threadIdx.x == 0) {
    float mx = lg[0];
    for (int i = 1; i < kE; i++) mx = fmaxf(mx, lg[i]);
    float pr[kE], s = 0.0f;
    for (int i = 0; i < kE; i++) { pr[i] = expf(lg[i] - mx); s += pr[i]; }
    for (int i = 0; i < kE; i++) pr[i] /= s;
    int i0 = 0;
    for (int i = 1; i < kE; i++) if (pr[i] > pr[i0]) i0 = i;
    int i1 = (i0 == 0) ? 1 : 0;
    for (int i = 0; i < kE; i++)
      if (i != i0 && pr[i] > pr[i1]) i1 = i;
    float sw = pr[i0] + pr[i1] + 1e-8f;
    topi[t * 2 + 0] = i0; topw[t * 2 + 0] = pr[i0] / sw;
    topi[t * 2 + 1] = i1; topw[t * 2 + 1] = pr[i1] / sw;
    atomicAdd(&cnt[i0], 1); atomicAdd(&cnt[i1], 1);
    atomicAdd(&f_cnt[i0], 1.0f); atomicAdd(&f_cnt[i1], 1.0f);
    for (int i = 0; i < kE; i++) atomicAdd(&p_sum[i], pr[i]);
  }
}

__global__ void k_finalize(const int* cnt, const float* wsum,
                           const float* p_sum, const float* f_cnt, int* off,
                           float* wm, int* tileExpert, float* aux_out) {
  __shared__ int s_off[kE + 1];
  int tid = threadIdx.x;
  if (tid == 0) {
    int run = 0;
    for (int e = 0; e < kE; e++) {
      off[e] = run; s_off[e] = run;
      run += (cnt[e] + 31) & ~31;
    }
    off[kE] = run; s_off[kE] = run;
    float aux = 0.0f;
    for (int e = 0; e < kE; e++)
      aux += (f_cnt[e] / (float)(kN * 2)) * (p_sum[e] / (float)kN);
    aux_out[0] = aux * (float)kE;
  }
  if (tid < 2 * kE) {
    float mean = wsum[tid] / (float)((size_t)kF * kD);
    wm[tid] = fmaxf(mean, 1e-5f);
  }
  __syncthreads();
  for (int t = tid; t < kMaxTiles; t += blockDim.x) {
    int r = t * 16, e = 0;
    for (int i = 0; i < kE; i++)
      if (r >= s_off[i] && r < s_off[i + 1]) e = i;
    tileExpert[t] = e;
  }
}

__global__ void k_scatter(const int* topi, const float* topw, const int* off,
                          int* fill, int* perm, float* pweight) {
  int t = blockIdx.x * blockDim.x + threadIdx.x;
  if (t >= kN) return;
  for (int k = 0; k < 2; k++) {
    int e = topi[t * 2 + k];
    int pos = off[e] + atomicAdd(&fill[e], 1);
    perm[pos] = t;
    pweight[pos] = topw[t * 2 + k];
  }
}

__global__ void k_quant_w(const float* w1, const float* w2, const float* wm,
                          int8_t* qw1, int8_t* qw2) {
  int e = blockIdx.y, z = blockIdx.z;
  const float4* src = (const float4*)((z ? w2 : w1) + (size_t)e * kF * kD);
  int* dst = (int*)((z ? qw2 : qw1) + (size_t)e * kF * kD);
  float qs = 1.0f / wm[z * kE + e];
  size_t n4 = (size_t)kF * kD / 4;
  for (size_t i = (size_t)blockIdx.x * blockDim.x + threadIdx.x; i < n4;
       i += (size_t)gridDim.x * blockDim.x) {
    float4 v = src[i];
    int q0 = (int)fmaxf(-1.0f, fminf(1.0f, rintf(v.x * qs)));
    int q1 = (int)fmaxf(-1.0f, fminf(1.0f, rintf(v.y * qs)));
    int q2 = (int)fmaxf(-1.0f, fminf(1.0f, rintf(v.z * qs)));
    int q3 = (int)fmaxf(-1.0f, fminf(1.0f, rintf(v.w * qs)));
    dst[i] = (q0 & 0xFF) | ((q1 & 0xFF) << 8) | ((q2 & 0xFF) << 16) | (q3 << 24);
  }
}

__global__ void k_actquant1(const float* x, const float* g1, const int* perm,
                            const int* tileExpert, int8_t* qa, float* scale1) {
  int r = blockIdx.x, tid = threadIdx.x;
  int e = tileExpert[r >> 4];
  int token = perm[r];
  int* qrow = (int*)(qa + (size_t)r * kD);
  __shared__ float red[256];
  if (token < 0) {
    qrow[tid] = 0;
    if (tid == 0) scale1[r] = 0.0f;
    return;
  }
  const float* xr = x + (size_t)token * kD;
  const float* gr = g1 + (size_t)e * kD;
  float v[4], ss = 0.0f;
  for (int j = 0; j < 4; j++) { v[j] = xr[tid * 4 + j]; ss += v[j] * v[j]; }
  red[tid] = ss; __syncthreads();
  for (int w = 128; w > 0; w >>= 1) {
    if (tid < w) red[tid] += red[tid + w];
    __syncthreads();
  }
  float rms = rsqrtf(red[0] / (float)kD + kRmsEps);
  __syncthreads();
  float mx = 0.0f;
  for (int j = 0; j < 4; j++) {
    v[j] = v[j] * gr[tid * 4 + j] * rms;
    mx = fmaxf(mx, fabsf(v[j]));
  }
  red[tid] = mx; __syncthreads();
  for (int w = 128; w > 0; w >>= 1) {
    if (tid < w) red[tid] = fmaxf(red[tid], red[tid + w]);
    __syncthreads();
  }
  float amax = fmaxf(red[0], 1e-5f);
  float s = 127.0f / amax;
  int pk = 0;
  for (int j = 0; j < 4; j++) {
    int q = (int)fmaxf(-128.0f, fminf(127.0f, rintf(v[j] * s)));
    pk |= (q & 0xFF) << (8 * j);
  }
  qrow[tid] = pk;
  if (tid == 0) scale1[r] = amax / 127.0f;
}

__global__ void k_actquant2(const float* h, const float* g2,
                            const int* tileExpert, int8_t* qh, float* scale2) {
  int r = blockIdx.x, tid = threadIdx.x;
  int e = tileExpert[r >> 4];
  const float* hr = h + (size_t)r * kF;
  const float* gr = g2 + (size_t)e * kF;
  __shared__ float red[256];
  float v[16], ss = 0.0f;
  for (int j = 0; j < 16; j++) { v[j] = hr[tid * 16 + j]; ss += v[j] * v[j]; }
  red[tid] = ss; __syncthreads();
  for (int w = 128; w > 0; w >>= 1) {
    if (tid < w) red[tid] += red[tid + w];
    __syncthreads();
  }
  float rms = rsqrtf(red[0] / (float)kF + kRmsEps);
  __syncthreads();
  float mx = 0.0f;
  for (int j = 0; j < 16; j++) {
    v[j] = v[j] * gr[tid * 16 + j] * rms;
    mx = fmaxf(mx, fabsf(v[j]));
  }
  red[tid] = mx; __syncthreads();
  for (int w = 128; w > 0; w >>= 1) {
    if (tid < w) red[tid] = fmaxf(red[tid], red[tid + w]);
    __syncthreads();
  }
  float amax = fmaxf(red[0], 1e-5f);
  float s = 127.0f / amax;
  int* qrow = (int*)(qh + (size_t)r * kF);
  for (int j4 = 0; j4 < 4; j4++) {
    int pk = 0;
    for (int j = 0; j < 4; j++) {
      int q = (int)fmaxf(-128.0f, fminf(127.0f, rintf(v[j4 * 4 + j] * s)));
      pk |= (q & 0xFF) << (8 * j);
    }
    qrow[tid * 4 + j4] = pk;
  }
  if (tid == 0) scale2[r] = amax / 127.0f;
}

__global__ __launch_bounds__(128) void k_gemm1(const int8_t* qa,
    const int8_t* qw1, const float* scale1, const float* wm,
    const int* tileExpert, float* h) {
  int wv = threadIdx.x >> 5, lane = threadIdx.x & 31;
  int m = lane & 15, half = lane >> 4;
  int mgrp = blockIdx.y;
  int e = tileExpert[mgrp * 2];
  int ncolbase = blockIdx.x * 256 + wv * 64;
  int row0 = mgrp * 32 + m;
  const int8_t* qwE = qw1 + (size_t)e * kF * kD;
  v8i acc[2][4];
#pragma unroll
  for (int mi = 0; mi < 2; mi++)
#pragma unroll
    for (int j = 0; j < 4; j++) acc[mi][j] = (v8i){0, 0, 0, 0, 0, 0, 0, 0};
  for (int k0 = 0; k0 < kD; k0 += 64) {
    v8i a0 = load_a_tile(qa, row0, kD, k0, half);
    v8i a1 = load_a_tile(qa, row0 + 16, kD, k0, half);
#pragma unroll
    for (int j = 0; j < 4; j++) {
      v8i b = load_b_tile(qwE, ncolbase + j * 16 + m, kD, k0, half);
      acc[0][j] = __builtin_amdgcn_wmma_i32_16x16x64_iu8(true, a0, true, b,
                                                         acc[0][j], false, false);
      acc[1][j] = __builtin_amdgcn_wmma_i32_16x16x64_iu8(true, a1, true, b,
                                                         acc[1][j], false, false);
    }
  }
  float wmE = wm[e];
#pragma unroll
  for (int mi = 0; mi < 2; mi++) {
#pragma unroll
    for (int i = 0; i < 8; i++) {
      int mr = mgrp * 32 + mi * 16 + half * 8 + i;
      float rs = scale1[mr] * wmE;
#pragma unroll
      for (int j = 0; j < 4; j++) {
        float val = (float)acc[mi][j][i] * rs;
        h[(size_t)mr * kF + ncolbase + j * 16 + m] = gelu_tanh(val);
      }
    }
  }
}

__global__ __launch_bounds__(128) void k_gemm2(const int8_t* qh,
    const int8_t* qw2, const float* scale2, const float* wm,
    const int* tileExpert, const int* perm, const float* pweight, float* out) {
  int wv = threadIdx.x >> 5, lane = threadIdx.x & 31;
  int m = lane & 15, half = lane >> 4;
  int mgrp = blockIdx.y;
  int e = tileExpert[mgrp * 2];
  int ncolbase = blockIdx.x * 256 + wv * 64;
  int row0 = mgrp * 32 + m;
  const int8_t* qwE = qw2 + (size_t)e * kD * kF;
  v8i acc[2][4];
#pragma unroll
  for (int mi = 0; mi < 2; mi++)
#pragma unroll
    for (int j = 0; j < 4; j++) acc[mi][j] = (v8i){0, 0, 0, 0, 0, 0, 0, 0};
  for (int k0 = 0; k0 < kF; k0 += 64) {
    v8i a0 = load_a_tile(qh, row0, kF, k0, half);
    v8i a1 = load_a_tile(qh, row0 + 16, kF, k0, half);
#pragma unroll
    for (int j = 0; j < 4; j++) {
      v8i b = load_b_tile(qwE, ncolbase + j * 16 + m, kF, k0, half);
      acc[0][j] = __builtin_amdgcn_wmma_i32_16x16x64_iu8(true, a0, true, b,
                                                         acc[0][j], false, false);
      acc[1][j] = __builtin_amdgcn_wmma_i32_16x16x64_iu8(true, a1, true, b,
                                                         acc[1][j], false, false);
    }
  }
  float wmE = wm[kE + e];
#pragma unroll
  for (int mi = 0; mi < 2; mi++) {
#pragma unroll
    for (int i = 0; i < 8; i++) {
      int mr = mgrp * 32 + mi * 16 + half * 8 + i;
      int token = perm[mr];
      if (token >= 0) {
        float rs = scale2[mr] * wmE * pweight[mr];
#pragma unroll
        for (int j = 0; j < 4; j++) {
          float val = (float)acc[mi][j][i] * rs;
          atomicAdd(&out[(size_t)token * kD + ncolbase + j * 16 + m], val);
        }
      }
    }
  }
}

extern "C" void kernel_launch(void* const* d_in, const int* in_sizes, int n_in,
                              void* d_out, int out_size, void* d_ws,
                              size_t ws_size, hipStream_t stream) {
  (void)in_sizes; (void)n_in; (void)out_size; (void)ws_size;
  const float* x  = (const float*)d_in[0];
  const float* rw = (const float*)d_in[1];
  const float* w1 = (const float*)d_in[2];
  const float* g1 = (const float*)d_in[3];
  const float* w2 = (const float*)d_in[4];
  const float* g2 = (const float*)d_in[5];
  float* out = (float*)d_out;
  float* aux = out + (size_t)kN * kD;

  char* base = (char*)d_ws;
  size_t off = 0;
  auto carve = [&](size_t bytes) {
    void* p = base + off;
    off = (off + bytes + 255) & ~(size_t)255;
    return p;
  };
  int8_t* qw1    = (int8_t*)carve((size_t)kE * kF * kD);
  int8_t* qw2    = (int8_t*)carve((size_t)kE * kF * kD);
  int8_t* qa     = (int8_t*)carve((size_t)kMaxRows * kD);
  int8_t* qh     = (int8_t*)carve((size_t)kMaxRows * kF);
  float*  h      = (float*) carve((size_t)kMaxRows * kF * sizeof(float));
  float*  scale1 = (float*) carve(kMaxRows * sizeof(float));
  float*  scale2 = (float*) carve(kMaxRows * sizeof(float));
  int*    perm   = (int*)   carve(kMaxRows * sizeof(int));
  float*  pwgt   = (float*) carve(kMaxRows * sizeof(float));
  int*    topi   = (int*)   carve(kN * 2 * sizeof(int));
  float*  topw   = (float*) carve(kN * 2 * sizeof(float));
  int*    cnt    = (int*)   carve(kE * sizeof(int));
  int*    fill   = (int*)   carve(kE * sizeof(int));
  int*    offs   = (int*)   carve((kE + 1) * sizeof(int));
  float*  wsum   = (float*) carve(2 * kE * sizeof(float));
  float*  wm     = (float*) carve(2 * kE * sizeof(float));
  float*  p_sum  = (float*) carve(kE * sizeof(float));
  float*  f_cnt  = (float*) carve(kE * sizeof(float));
  int*    tileE  = (int*)   carve(kMaxTiles * sizeof(int));

  k_zero_out<<<((size_t)kN * kD + 255) / 256, 256, 0, stream>>>(out);
  k_init_misc<<<(kMaxRows + 255) / 256, 256, 0, stream>>>(
      perm, pwgt, cnt, fill, p_sum, f_cnt, wsum);
  k_weight_sum<<<dim3(256, kE, 2), 256, 0, stream>>>(w1, w2, wsum);
  k_router<<<kN, 512, 0, stream>>>(x, rw, topi, topw, cnt, p_sum, f_cnt);
  k_finalize<<<1, 256, 0, stream>>>(cnt, wsum, p_sum, f_cnt, offs, wm, tileE,
                                    aux);
  k_scatter<<<(kN + 255) / 256, 256, 0, stream>>>(topi, topw, offs, fill,
                                                  perm, pwgt);
  k_quant_w<<<dim3(1024, kE, 2), 256, 0, stream>>>(w1, w2, wm, qw1, qw2);
  k_actquant1<<<kMaxRows, 256, 0, stream>>>(x, g1, perm, tileE, qa, scale1);
  k_gemm1<<<dim3(kF / 256, kMGroups), 128, 0, stream>>>(qa, qw1, scale1, wm,
                                                        tileE, h);
  k_actquant2<<<kMaxRows, 256, 0, stream>>>(h, g2, tileE, qh, scale2);
  k_gemm2<<<dim3(kD / 256, kMGroups), 128, 0, stream>>>(
      qh, qw2, scale2, wm, tileE, perm, pwgt, out);
}
